// RBM_20160576487673
// MI455X (gfx1250) — matrix operand order repaired and checked by reading, not compiled, not run on hardware
//
#include <hip/hip_runtime.h>
#include <stdint.h>

// ---------------------------------------------------------------------------
// RBM Gibbs chain on MI455X (gfx1250, wave32, WMMA + async global->LDS).
//   ph = sigmoid(v @ W^T + b_h)   : [B,V] x [V,H] -> [B,H]
//   pv = sigmoid(h @ W   + b_v)   : [B,H] x [H,V] -> [B,V]
// bf16 inputs / f32 accum via V_WMMA_F32_16X16X32_BF16.
// B operand consumed row-major in [N,K] form:
//   - ph GEMM: [N,K] = [H,V] = W (bf16 copy)
//   - pv GEMM: [N,K] = [V,H] = W^T (bf16, built once)
// 128x128 block tiles double-buffered in LDS via GLOBAL_LOAD_ASYNC_TO_LDS_B128
// (ASYNCcnt) so each weight row is fetched from L2 once per block, not once
// per wave. Weights (16MB bf16 total) stay L2-resident across all 40 GEMMs.
// ---------------------------------------------------------------------------

typedef __bf16 bf16;
typedef bf16  v8bf  __attribute__((ext_vector_type(8)));
typedef bf16  v16bf __attribute__((ext_vector_type(16)));
typedef float v8f   __attribute__((ext_vector_type(8)));
typedef int   v4i   __attribute__((ext_vector_type(4)));

#define B_DIM 4096
#define V_DIM 4096
#define H_DIM 1024
#define NUM_GIB 20

#define M_BLK 128
#define N_BLK 128
#define KB    64          // K elements staged per pipeline stage
#define SROW  72          // padded LDS row stride (elems): 144B = 36 banks,
                          // conflict-free for b128 reads by 16-lane row groups

// ---- gfx1250 async copy: global -> LDS (16 bytes), tracked by ASYNCcnt ----
// Probe-confirmed signature: (int4* gaddr, int4* ldsaddr, imm offset, imm cpol)
__device__ __forceinline__ void async_cp_b128(const bf16* g, bf16* l) {
#if __has_builtin(__builtin_amdgcn_global_load_async_to_lds_b128)
  __builtin_amdgcn_global_load_async_to_lds_b128(
      (v4i*)(uintptr_t)g, (v4i*)(uintptr_t)l, 0, 0);
#else
  asm volatile("global_load_async_to_lds_b128 %0, %1, off"
               :: "v"((uint32_t)(uintptr_t)l), "v"(g) : "memory");
#endif
}

__device__ __forceinline__ void wait_async0() {
#if __has_builtin(__builtin_amdgcn_s_wait_asynccnt)
  __builtin_amdgcn_s_wait_asynccnt(0);
#else
  asm volatile("s_wait_asynccnt 0x0" ::: "memory");
#endif
}

// Counter-based hash RNG -> uniform [0,1)
__device__ __forceinline__ float rng01(uint32_t x) {
  x ^= x >> 17; x *= 0xed5ad4bbu;
  x ^= x >> 11; x *= 0xac4c1b51u;
  x ^= x >> 15; x *= 0x31848babu;
  x ^= x >> 14;
  return (float)(x >> 8) * (1.0f / 16777216.0f);
}

__device__ __forceinline__ float sigmoidf_fast(float x) {
  return 1.0f / (1.0f + __expf(-x));
}

// ---------------------------------------------------------------------------
// Fused GEMM + bias + sigmoid + Bernoulli sampling.
// A  : [M,K] bf16 row-major ; Bs : [N,K] bf16 row-major.
// Block: 256 threads = 8 waves (4 M x 2 N), each wave owns 32x64 of a 128x128
// block tile; K pipelined in KB=64 chunks, double-buffered in LDS.
// ---------------------------------------------------------------------------
__global__ __launch_bounds__(256)
void rbm_gemm_sigmoid_sample(const bf16* __restrict__ A,
                             const bf16* __restrict__ Bs,
                             const float* __restrict__ bias,   // [N]
                             bf16*  __restrict__ out_s_bf16,   // sampled {0,1} (nullable)
                             float* __restrict__ out_p_f32,    // probabilities (nullable)
                             float* __restrict__ out_s_f32,    // sampled fp32  (nullable)
                             int M, int N, int K, uint32_t seed)
{
  __shared__ bf16 sA[2][M_BLK * SROW];   // 2 x 18KB
  __shared__ bf16 sB[2][N_BLK * SROW];   // 2 x 18KB

  const int tid  = threadIdx.x;
  const int lane = tid & 31;
  const int wv   = tid >> 5;     // wave in block: 0..7
  const int wm   = wv >> 1;      // 0..3  -> 32-row slice
  const int wn   = wv & 1;       // 0..1  -> 64-col slice
  const int half = lane >> 4;    // 0: lanes 0-15, 1: lanes 16-31
  const int l16  = lane & 15;

  const int nblk = N / N_BLK;
  const int bm = blockIdx.x / nblk;
  const int bn = blockIdx.x - bm * nblk;
  const int m0 = bm * M_BLK;
  const int n0 = bn * N_BLK;

  // --- async-copy assignment: thread -> (row cr, 64-byte half ch) ---------
  // tile row = KB elems = 128B = 8 x b128; 2 threads/row, 4 b128 each.
  const int cr = tid >> 1;          // 0..127
  const int ch = tid & 1;           // 0,1
  const bf16* gA = A  + (size_t)(m0 + cr) * K + ch * 32;
  const bf16* gB = Bs + (size_t)(n0 + cr) * K + ch * 32;
  bf16* lA0 = &sA[0][cr * SROW + ch * 32];
  bf16* lA1 = &sA[1][cr * SROW + ch * 32];
  bf16* lB0 = &sB[0][cr * SROW + ch * 32];
  bf16* lB1 = &sB[1][cr * SROW + ch * 32];

  auto issue = [&](int buf, int k) {
    bf16* la = buf ? lA1 : lA0;
    bf16* lb = buf ? lB1 : lB0;
    #pragma unroll
    for (int j = 0; j < 4; ++j) {
      async_cp_b128(gA + k + j * 8, la + j * 8);
      async_cp_b128(gB + k + j * 8, lb + j * 8);
    }
  };

  // --- accumulators --------------------------------------------------------
  const v8f vzero = {0.f,0.f,0.f,0.f,0.f,0.f,0.f,0.f};
  v8f acc[2][4];
  #pragma unroll
  for (int mt = 0; mt < 2; ++mt)
    #pragma unroll
    for (int nt = 0; nt < 4; ++nt)
      acc[mt][nt] = vzero;

  // fragment base rows in LDS (per lane)
  const int a_row = (wm * 32 + l16) * SROW;   // + mt*16*SROW
  const int b_row = (wn * 64 + l16) * SROW;   // + nt*16*SROW
  const int a_off = 8  * half;   // A elems 0..7: K=8*half+e ; 8..15: +16
  const int b_off = 8  * half;   // B elems 0..7: K=8*half+e ; 8..15: +16

  const int KT = K / KB;
  issue(0, 0);

  for (int kt = 0; kt < KT; ++kt) {
    wait_async0();        // own wave's async copies done
    __syncthreads();      // => every wave's copies done; prev-prev buffer free
    if (kt + 1 < KT) issue((kt + 1) & 1, (kt + 1) * KB);

    const bf16* cA = sA[kt & 1];
    const bf16* cB = sB[kt & 1];

    #pragma unroll
    for (int ks = 0; ks < KB; ks += 32) {
      union { v16bf v; v8bf h[2]; } a[2], b[4];
      #pragma unroll
      for (int mt = 0; mt < 2; ++mt) {
        const bf16* p = cA + a_row + mt * 16 * SROW + ks;
        a[mt].h[0] = *(const v8bf*)(p + a_off);
        a[mt].h[1] = *(const v8bf*)(p + a_off + 16);
      }
      #pragma unroll
      for (int nt = 0; nt < 4; ++nt) {
        const bf16* p = cB + b_row + nt * 16 * SROW + ks;
        b[nt].h[0] = *(const v8bf*)(p + b_off);
        b[nt].h[1] = *(const v8bf*)(p + b_off + 16);
      }
      #pragma unroll
      for (int mt = 0; mt < 2; ++mt)
        #pragma unroll
        for (int nt = 0; nt < 4; ++nt)
          acc[mt][nt] = __builtin_amdgcn_wmma_f32_16x16x32_bf16(
              false, a[mt].v, false, b[nt].v,
              (short)0, acc[mt][nt], false, false);
    }
  }

  // --- epilogue: C/D layout -> VGPR r: lanes 0-15 M=r, lanes 16-31 M=8+r ---
  #pragma unroll
  for (int mt = 0; mt < 2; ++mt) {
    #pragma unroll
    for (int nt = 0; nt < 4; ++nt) {
      const int col = n0 + wn * 64 + nt * 16 + l16;
      const float bcol = bias[col];
      #pragma unroll
      for (int r = 0; r < 8; ++r) {
        const int row = m0 + wm * 32 + mt * 16 + half * 8 + r;
        const float p = sigmoidf_fast(acc[mt][nt][r] + bcol);
        const uint32_t idx = (uint32_t)(row * N + col);   // < 2^24, exact
        const float u = rng01(idx * 0x9E3779B9u + seed);
        const float s = (u < p) ? 1.0f : 0.0f;
        const size_t o = (size_t)row * (size_t)N + (size_t)col;
        if (out_s_bf16) out_s_bf16[o] = (bf16)s;
        if (out_p_f32)  out_p_f32[o]  = p;
        if (out_s_f32)  out_s_f32[o]  = s;
      }
    }
  }
}

// fp32 -> bf16 elementwise
__global__ void cvt_f32_bf16(const float* __restrict__ in, bf16* __restrict__ out, int n) {
  int i = blockIdx.x * blockDim.x + threadIdx.x;
  if (i < n) out[i] = (bf16)in[i];
}

// W [H,V] fp32 -> WT [V,H] bf16 (output-coalesced)
__global__ void transpose_w_bf16(const float* __restrict__ W, bf16* __restrict__ WT) {
  int i = blockIdx.x * blockDim.x + threadIdx.x;
  if (i < V_DIM * H_DIM) {
    int v = i >> 10;            // / H_DIM
    int h = i & (H_DIM - 1);
    WT[i] = (bf16)W[(size_t)h * V_DIM + v];
  }
}

extern "C" void kernel_launch(void* const* d_in, const int* in_sizes, int n_in,
                              void* d_out, int out_size, void* d_ws, size_t ws_size,
                              hipStream_t stream) {
  const float* v_in   = (const float*)d_in[0];   // [B,V]
  const float* weight = (const float*)d_in[1];   // [H,V]
  const float* bias_v = (const float*)d_in[2];   // [V]
  const float* bias_h = (const float*)d_in[3];   // [H]
  // d_in[4] = num_gib (device scalar) -> compile-time NUM_GIB=20

  // Workspace layout (needs 56MB):
  char* ws = (char*)d_ws;
  bf16* Wb  = (bf16*)ws;                                              // [H,V]  8MB
  bf16* WTb = (bf16*)(ws + (size_t)H_DIM * V_DIM * 2);                // [V,H]  8MB
  bf16* vb  = (bf16*)(ws + (size_t)H_DIM * V_DIM * 4);                // [B,V] 32MB
  bf16* hb  = (bf16*)(ws + (size_t)H_DIM * V_DIM * 4
                         + (size_t)B_DIM * V_DIM * 2);                // [B,H]  8MB

  float* out_v  = (float*)d_out;                        // v_gibb  [B,V]
  float* out_pv = out_v + (size_t)B_DIM * V_DIM;        // pv_gibb [B,V]

  // --- one-time prep (per call; deterministic) ---
  {
    int n = H_DIM * V_DIM;
    cvt_f32_bf16<<<(n + 255) / 256, 256, 0, stream>>>(weight, Wb, n);
    transpose_w_bf16<<<(n + 255) / 256, 256, 0, stream>>>(weight, WTb);
    n = B_DIM * V_DIM;
    cvt_f32_bf16<<<(n + 255) / 256, 256, 0, stream>>>(v_in, vb, n);
  }

  auto launch_gemm = [&](const bf16* A, const bf16* Bs, const float* bias,
                         bf16* os, float* op, float* osf,
                         int M, int N, int K, uint32_t seed) {
    int blocks = (M / M_BLK) * (N / N_BLK);
    rbm_gemm_sigmoid_sample<<<blocks, 256, 0, stream>>>(
        A, Bs, bias, os, op, osf, M, N, K, seed);
  };

  // ph0 = sigmoid(v @ W^T + b_h); sample h0
  launch_gemm(vb, Wb, bias_h, hb, nullptr, nullptr,
              B_DIM, H_DIM, V_DIM, 0x13572468u);

  for (int t = 0; t < NUM_GIB; ++t) {
    const bool last = (t == NUM_GIB - 1);
    // pv = sigmoid(h @ W + b_v); sample v_g
    launch_gemm(hb, WTb, bias_v,
                last ? nullptr : vb,
                last ? out_pv  : nullptr,
                last ? out_v   : nullptr,
                B_DIM, V_DIM, H_DIM, 0xA5A50000u + 2u * t);
    if (!last) {
      // ph = sigmoid(v_g @ W^T + b_h); sample h
      launch_gemm(vb, Wb, bias_h, hb, nullptr, nullptr,
                  B_DIM, H_DIM, V_DIM, 0x5A5A0000u + 2u * t + 1u);
    }
  }
}
